// CausalSelfAttention_61521111548240
// MI455X (gfx1250) — matrix operand order repaired and checked by reading, not compiled, not run on hardware
//
#include <hip/hip_runtime.h>
#include <hip/hip_bf16.h>

// CausalSelfAttention for T=2048, C=2048, H=16, hd=128 on gfx1250 (MI455X).
// GEMMs: v_wmma_f32_16x16x32_bf16 with async-double-buffered LDS staging
// (global_load_async_to_lds_b128 / s_wait_asynccnt). Attention: flash-style
// single pass, WMMA for Q*K^T and P*V.

typedef __attribute__((ext_vector_type(16))) __bf16 v16bf;
typedef __attribute__((ext_vector_type(8)))  __bf16 v8bf;
typedef __attribute__((ext_vector_type(8)))  float  v8f;

#define TT 2048
#define CC 2048
#define HH 16
#define HD 128

// GEMM macro-tile: 64(M) x 256(N) per workgroup (8 waves, each 32x64), K step 32.
#define WG_M 64
#define WG_N 256
#define KSTEP 32
#define LDS_PITCH 40  // halfs per panel row (32 data + 8 pad) -> conflict-free banks
#define A_HALFS (WG_M * LDS_PITCH)            // 2560
#define B_HALFS (WG_N * LDS_PITCH)            // 10240
#define BUF_HALFS (A_HALFS + B_HALFS)         // 12800 halfs = 25 KB

// ---------- helpers ----------

__device__ __forceinline__ unsigned short f2bf_bits(float f) {
  unsigned u = __float_as_uint(f);
  unsigned r = u + 0x7FFFu + ((u >> 16) & 1u);   // round-to-nearest-even
  return (unsigned short)(r >> 16);
}

// A-fragment (16x32 MxK, bf16): lane L holds row m=L%16, K chunks at
// k_base + (L/16)*8 .. +7  and  k_base + 16 + (L/16)*8 .. +7   (ISA 7.12.2)
__device__ __forceinline__ v16bf load_a_frag(const unsigned short* base, int stride,
                                             int m_base, int k_base, int lane) {
  int m = m_base + (lane & 15);
  int h = lane >> 4;
  const unsigned short* row = base + (size_t)m * stride + k_base;
  union { v16bf v; v8bf h2[2]; } u;
  u.h2[0] = *(const v8bf*)(row + h * 8);
  u.h2[1] = *(const v8bf*)(row + 16 + h * 8);
  return u.v;
}

// B-fragment (32x16 KxN, bf16) where B[k][n] = W[n_base+n][k_base+k], W row-major:
// lane L holds column n=L%16, K = (L/16)*8 .. +7 and 16 + (L/16)*8 .. +7 (two 16B loads).
__device__ __forceinline__ v16bf load_b_frag(const unsigned short* base, int stride,
                                             int n_base, int k_base, int lane) {
  int n = n_base + (lane & 15);
  int k = k_base + (lane >> 4) * 8;
  const unsigned short* p = base + (size_t)n * stride + k;
  union { v16bf v; v8bf h2[2]; } u;
  u.h2[0] = *(const v8bf*)(p);
  u.h2[1] = *(const v8bf*)(p + 16);
  return u.v;
}

__device__ __forceinline__ v8f wmma_bf16(v16bf a, v16bf b, v8f c) {
  return __builtin_amdgcn_wmma_f32_16x16x32_bf16(false, a, false, b, (short)0, c,
                                                 false, false);
}

// Async-copy a (rows x 32-half) tile from global (stride ld halfs) into LDS
// (pitch LDS_PITCH halfs). 512B chunk = 8 rows; lane moves 16B. Distributed
// over the 8 waves of the block; tracked on ASYNCcnt.
__device__ __forceinline__ void async_tile(const unsigned short* __restrict__ g, int ld,
                                           int k0, unsigned lds_base_bytes, int rows,
                                           int wave, int lane) {
  int chunks = rows >> 3;
  for (int c = wave; c < chunks; c += 8) {
    int row = c * 8 + (lane >> 2);
    int kp  = (lane & 3) * 8;                       // half offset within the 32-K row
    const unsigned short* src = g + (size_t)row * ld + k0 + kp;
    unsigned dst = lds_base_bytes + (unsigned)(row * (LDS_PITCH * 2) + kp * 2);
    asm volatile("global_load_async_to_lds_b128 %0, %1, off"
                 :: "v"(dst), "v"((unsigned long long)(uintptr_t)src)
                 : "memory");
  }
}

// ---------- fp32 -> bf16 ----------

__global__ __launch_bounds__(256) void f32_to_bf16_kernel(const float* __restrict__ in,
                                                          unsigned short* __restrict__ out,
                                                          int n) {
  int i = blockIdx.x * blockDim.x + threadIdx.x;
  if (i < n) out[i] = f2bf_bits(in[i]);
}

// ---------- V transpose: vt[h][d][t] = qkv[t][4096 + h*128 + d] ----------

__global__ __launch_bounds__(256) void v_transpose_kernel(const unsigned short* __restrict__ qkv,
                                                          unsigned short* __restrict__ vt) {
  int i = blockIdx.x * blockDim.x + threadIdx.x;
  if (i >= TT * CC) return;
  int t = i / CC;
  int c = i % CC;
  vt[(size_t)c * TT + t] = qkv[(size_t)t * (3 * CC) + 2 * CC + c];
}

// ---------- WMMA GEMM: C = A (MxK) * B^T (B is NxK row-major) ----------
// 8 waves stage A(64x32) + B(256x32) panels in LDS via async loads, double
// buffered; each wave computes a 32x64 tile out of LDS.

template <bool OUT_BF16>
__global__ __launch_bounds__(256) void gemm_bf16(const unsigned short* __restrict__ A, int lda,
                                                 const unsigned short* __restrict__ B, int ldb,
                                                 void* __restrict__ Cp, int ldc,
                                                 int M, int N, int K) {
  __shared__ __align__(128) unsigned short smem[2 * BUF_HALFS];  // 50 KB

  int lane = threadIdx.x & 31;
  int wave = threadIdx.x >> 5;
  int tiles_n = N / WG_N;
  int m0 = (blockIdx.x / tiles_n) * WG_M;
  int n0 = (blockIdx.x % tiles_n) * WG_N;

  unsigned smem_base = (unsigned)(uintptr_t)(const void*)smem;  // LDS byte offset
  const unsigned abase0 = smem_base;
  const unsigned bbase0 = smem_base + A_HALFS * 2;
  const unsigned abase1 = smem_base + BUF_HALFS * 2;
  const unsigned bbase1 = abase1 + A_HALFS * 2;

  // prologue: buffer 0 (k=0), then buffer 1 (k=32) in flight
  async_tile(A + (size_t)m0 * lda, lda, 0, abase0, WG_M, wave, lane);
  async_tile(B + (size_t)n0 * ldb, ldb, 0, bbase0, WG_N, wave, lane);
  asm volatile("s_wait_asynccnt 0x0" ::: "memory");
  __syncthreads();
  async_tile(A + (size_t)m0 * lda, lda, KSTEP, abase1, WG_M, wave, lane);
  async_tile(B + (size_t)n0 * ldb, ldb, KSTEP, bbase1, WG_N, wave, lane);

  v8f acc[2][4];
#pragma unroll
  for (int mt = 0; mt < 2; ++mt)
#pragma unroll
    for (int nt = 0; nt < 4; ++nt)
      acc[mt][nt] = (v8f){0.f, 0.f, 0.f, 0.f, 0.f, 0.f, 0.f, 0.f};

  int wm = wave >> 2;        // 0..1 -> 32-row slice
  int wn = wave & 3;         // 0..3 -> 64-col slice

  for (int k0 = 0; k0 < K; k0 += KSTEP) {
    int cur = (k0 / KSTEP) & 1;
    const unsigned short* As = smem + (size_t)cur * BUF_HALFS;
    const unsigned short* Bs = As + A_HALFS;

    v16bf a0 = load_a_frag(As, LDS_PITCH, wm * 32, 0, lane);
    v16bf a1 = load_a_frag(As, LDS_PITCH, wm * 32 + 16, 0, lane);
#pragma unroll
    for (int nt = 0; nt < 4; ++nt) {
      v16bf b = load_b_frag(Bs, LDS_PITCH, wn * 64 + nt * 16, 0, lane);
      acc[0][nt] = wmma_bf16(a0, b, acc[0][nt]);
      acc[1][nt] = wmma_bf16(a1, b, acc[1][nt]);
    }

    __syncthreads();  // everyone done reading buffer `cur`
    if (k0 + 2 * KSTEP < K) {
      unsigned ab = cur ? abase1 : abase0;
      unsigned bb = cur ? bbase1 : bbase0;
      async_tile(A + (size_t)m0 * lda, lda, k0 + 2 * KSTEP, ab, WG_M, wave, lane);
      async_tile(B + (size_t)n0 * ldb, ldb, k0 + 2 * KSTEP, bb, WG_N, wave, lane);
      // 10 outstanding; async loads retire in order -> <=5 means the batch for
      // the *next* buffer has fully landed.
      asm volatile("s_wait_asynccnt 0x5" ::: "memory");
    } else {
      asm volatile("s_wait_asynccnt 0x0" ::: "memory");
    }
    __syncthreads();  // next buffer visible to all waves
  }

  // C f32 fragment: lane L holds column n=L%16, rows r + 8*(L/16), r=0..7
  int r_off = (lane >> 4) * 8;
  int cn = lane & 15;
#pragma unroll
  for (int mt = 0; mt < 2; ++mt)
#pragma unroll
    for (int nt = 0; nt < 4; ++nt)
#pragma unroll
      for (int r = 0; r < 8; ++r) {
        int row = m0 + wm * 32 + mt * 16 + r_off + r;
        int col = n0 + wn * 64 + nt * 16 + cn;
        float v = acc[mt][nt][r];
        if (OUT_BF16)
          ((unsigned short*)Cp)[(size_t)row * ldc + col] = f2bf_bits(v);
        else
          ((float*)Cp)[(size_t)row * ldc + col] = v;
      }
}

// ---------- flash attention: one wave handles (head h, 16 query rows) ----------

__global__ __launch_bounds__(32) void attn_fwd_kernel(const unsigned short* __restrict__ qkv,
                                                      const unsigned short* __restrict__ vt,
                                                      unsigned short* __restrict__ y) {
  __shared__ __align__(64) unsigned short pbuf[16 * 32];

  int lane = threadIdx.x;
  const int blocks_per_head = TT / 16;
  int h = blockIdx.x / blocks_per_head;
  int tb = (blockIdx.x % blocks_per_head) * 16;

  const int qcol = h * HD;
  const int kcol = CC + h * HD;
  const unsigned short* vth = vt + (size_t)h * HD * TT;

  // Q fragments for the 16x128 query tile (4 K-chunks of 32), loaded once.
  v16bf qf[4];
#pragma unroll
  for (int kc = 0; kc < 4; ++kc)
    qf[kc] = load_a_frag(qkv, 3 * CC, tb, qcol + kc * 32, lane);

  v8f acc_o[8];
#pragma unroll
  for (int nc = 0; nc < 8; ++nc)
    acc_o[nc] = (v8f){0.f, 0.f, 0.f, 0.f, 0.f, 0.f, 0.f, 0.f};

  float mrow[8], lrow[8];
#pragma unroll
  for (int r = 0; r < 8; ++r) { mrow[r] = -INFINITY; lrow[r] = 0.f; }

  const float scale = 0.08838834764831845f;  // 1/sqrt(128)
  int r_off = (lane >> 4) * 8;
  int cn = lane & 15;

  for (int jb = 0; jb <= tb + 15; jb += 32) {
    // S = Q * K^T for 32 keys (two 16-wide key blocks)
    v8f s0 = (v8f){0.f, 0.f, 0.f, 0.f, 0.f, 0.f, 0.f, 0.f};
    v8f s1 = (v8f){0.f, 0.f, 0.f, 0.f, 0.f, 0.f, 0.f, 0.f};
#pragma unroll
    for (int kc = 0; kc < 4; ++kc) {
      v16bf kf0 = load_b_frag(qkv, 3 * CC, jb, kcol + kc * 32, lane);
      v16bf kf1 = load_b_frag(qkv, 3 * CC, jb + 16, kcol + kc * 32, lane);
      s0 = wmma_bf16(qf[kc], kf0, s0);
      s1 = wmma_bf16(qf[kc], kf1, s1);
    }

    // online softmax (rows r+8*(lane/16) live in 16-lane halves -> shfl_xor<16)
    float sfr[8];
#pragma unroll
    for (int r = 0; r < 8; ++r) {
      int t = tb + r_off + r;
      float e0 = s0[r] * scale;
      float e1 = s1[r] * scale;
      e0 = (jb + cn <= t) ? e0 : -INFINITY;
      e1 = (jb + 16 + cn <= t) ? e1 : -INFINITY;
      float mx = fmaxf(e0, e1);
      mx = fmaxf(mx, __shfl_xor(mx, 1, 32));
      mx = fmaxf(mx, __shfl_xor(mx, 2, 32));
      mx = fmaxf(mx, __shfl_xor(mx, 4, 32));
      mx = fmaxf(mx, __shfl_xor(mx, 8, 32));
      float m_new = fmaxf(mrow[r], mx);
      float sf = __expf(mrow[r] - m_new);
      float p0 = __expf(e0 - m_new);
      float p1 = __expf(e1 - m_new);
      float ps = p0 + p1;
      ps += __shfl_xor(ps, 1, 32);
      ps += __shfl_xor(ps, 2, 32);
      ps += __shfl_xor(ps, 4, 32);
      ps += __shfl_xor(ps, 8, 32);
      lrow[r] = lrow[r] * sf + ps;
      mrow[r] = m_new;
      sfr[r] = sf;
      // stage P (C-fragment layout) to LDS row-major [16][32]
      pbuf[(r_off + r) * 32 + cn]      = f2bf_bits(p0);
      pbuf[(r_off + r) * 32 + 16 + cn] = f2bf_bits(p1);
    }

    // rescale O by per-row correction factor
#pragma unroll
    for (int nc = 0; nc < 8; ++nc)
#pragma unroll
      for (int r = 0; r < 8; ++r)
        acc_o[nc][r] *= sfr[r];

    __syncthreads();

    // reload P in A-fragment layout from LDS
    union { v16bf v; v8bf h2[2]; } pu;
    {
      int m = lane & 15;
      int hh = lane >> 4;
      pu.h2[0] = *(const v8bf*)&pbuf[m * 32 + hh * 8];
      pu.h2[1] = *(const v8bf*)&pbuf[m * 32 + 16 + hh * 8];
    }

    // O += P * V   (V^T stored (hd, T): contiguous B-fragments)
#pragma unroll
    for (int nc = 0; nc < 8; ++nc) {
      v16bf vf = load_b_frag(vth, TT, nc * 16, jb, lane);
      acc_o[nc] = wmma_bf16(pu.v, vf, acc_o[nc]);
    }
    __syncthreads();
  }

  // normalize and store y (bf16, (T, C) row-major for the output projection)
#pragma unroll
  for (int r = 0; r < 8; ++r) {
    float inv = 1.0f / lrow[r];
    int t = tb + r_off + r;
#pragma unroll
    for (int nc = 0; nc < 8; ++nc)
      y[(size_t)t * CC + h * HD + nc * 16 + cn] = f2bf_bits(acc_o[nc][r] * inv);
  }
}

// ---------- launch ----------

extern "C" void kernel_launch(void* const* d_in, const int* in_sizes, int n_in,
                              void* d_out, int out_size, void* d_ws, size_t ws_size,
                              hipStream_t stream) {
  const float* x      = (const float*)d_in[0];   // (T, C)
  const float* w_attn = (const float*)d_in[1];   // (3C, C)
  const float* w_proj = (const float*)d_in[2];   // (C, C)
  float* out = (float*)d_out;                    // (T, C)

  unsigned short* ws = (unsigned short*)d_ws;
  unsigned short* xb  = ws;                                   // T*C
  unsigned short* wab = xb  + (size_t)TT * CC;                // 3C*C
  unsigned short* wpb = wab + (size_t)3 * CC * CC;            // C*C
  unsigned short* qkv = wpb + (size_t)CC * CC;                // T*3C (q|k|v bf16)
  unsigned short* vt  = qkv + (size_t)TT * 3 * CC;            // (H, hd, T) = C*T
  unsigned short* yb  = vt  + (size_t)CC * TT;                // T*C

  // 1) convert inputs to bf16
  {
    int n = TT * CC;
    f32_to_bf16_kernel<<<(n + 255) / 256, 256, 0, stream>>>(x, xb, n);
    n = 3 * CC * CC;
    f32_to_bf16_kernel<<<(n + 255) / 256, 256, 0, stream>>>(w_attn, wab, n);
    n = CC * CC;
    f32_to_bf16_kernel<<<(n + 255) / 256, 256, 0, stream>>>(w_proj, wpb, n);
  }

  // 2) qkv = x @ w_attn^T  (bf16 out)
  {
    int blocks = (TT / WG_M) * ((3 * CC) / WG_N);   // 32 * 24 = 768
    gemm_bf16<true><<<blocks, 256, 0, stream>>>(xb, CC, wab, CC,
                                                (void*)qkv, 3 * CC,
                                                TT, 3 * CC, CC);
  }

  // 3) v -> v^T per head
  {
    int n = TT * CC;
    v_transpose_kernel<<<(n + 255) / 256, 256, 0, stream>>>(qkv, vt);
  }

  // 4) flash attention
  attn_fwd_kernel<<<HH * (TT / 16), 32, 0, stream>>>(qkv, vt, yb);

  // 5) out = y @ w_proj^T  (fp32 out)
  {
    int blocks = (TT / WG_M) * (CC / WG_N);         // 32 * 8 = 256
    gemm_bf16<false><<<blocks, 256, 0, stream>>>(yb, CC, wpb, CC,
                                                 (void*)out, CC,
                                                 TT, CC, CC);
  }
}
